// RNN_77137612636512
// MI455X (gfx1250) — matrix operand order repaired and checked by reading, not compiled, not run on hardware
//
#include <hip/hip_runtime.h>
#include <hip/hip_bf16.h>

typedef __attribute__((ext_vector_type(16))) _Float16 v16h;
typedef __attribute__((ext_vector_type(8)))  _Float16 v8h;
typedef __attribute__((ext_vector_type(8)))  float    v8f;
typedef __attribute__((ext_vector_type(4)))  float    v4f;
typedef __attribute__((ext_vector_type(4)))  int      i32x4;

#define T_STEPS 512
#define BATCH   64
#define IN_DIM  1024
#define HID     1024
#define KP      1040          // padded LDS column stride (f16 elems)
#define NBLK2   16            // persistent blocks in recurrent phase (HID/64)

// ---- CDNA5 feature guards ---------------------------------------------------
#if defined(__has_builtin)
#if __has_builtin(__builtin_amdgcn_global_store_async_from_lds_b128)
#define HAVE_ASYNC_ST 1
#endif
#if __has_builtin(__builtin_amdgcn_tanhf)
#define FAST_TANH(x) __builtin_amdgcn_tanhf(x)
#elif __has_builtin(__builtin_amdgcn_tanh_f32)
#define FAST_TANH(x) __builtin_amdgcn_tanh_f32(x)
#endif
#endif
#ifndef FAST_TANH
#define FAST_TANH(x) tanhf(x)
#endif

__device__ __forceinline__ void wait_asynccnt0() {
#if defined(__has_builtin) && __has_builtin(__builtin_amdgcn_s_wait_asynccnt)
  __builtin_amdgcn_s_wait_asynccnt(0);
#else
  asm volatile("s_wait_asynccnt 0" ::: "memory");
#endif
}

// ---------------------------------------------------------------------------
// init: zero both f16 state buffers and the grid-barrier words
// ---------------------------------------------------------------------------
__global__ void rnn_init(_Float16* __restrict__ s0, _Float16* __restrict__ s1,
                         unsigned* __restrict__ bar) {
  int idx = blockIdx.x * blockDim.x + threadIdx.x;
  int n = BATCH * HID;
  for (int i = idx; i < n; i += gridDim.x * blockDim.x) {
    s0[i] = (_Float16)0.0f;
    s1[i] = (_Float16)0.0f;
  }
  if (idx == 0) { bar[0] = 0u; bar[1] = 0u; }
}

// ---------------------------------------------------------------------------
// device-scope barrier for the persistent recurrent kernel
// ---------------------------------------------------------------------------
__device__ __forceinline__ void grid_barrier(unsigned* bar, unsigned nblk) {
  __syncthreads();
  if (threadIdx.x == 0) {
    __threadfence();  // make state stores visible device-wide
    unsigned g = __hip_atomic_load(&bar[1], __ATOMIC_ACQUIRE, __HIP_MEMORY_SCOPE_AGENT);
    unsigned old = __hip_atomic_fetch_add(&bar[0], 1u, __ATOMIC_ACQ_REL, __HIP_MEMORY_SCOPE_AGENT);
    if (old == nblk - 1u) {
      __hip_atomic_store(&bar[0], 0u, __ATOMIC_RELAXED, __HIP_MEMORY_SCOPE_AGENT);
      __hip_atomic_fetch_add(&bar[1], 1u, __ATOMIC_ACQ_REL, __HIP_MEMORY_SCOPE_AGENT);
    } else {
      while (__hip_atomic_load(&bar[1], __ATOMIC_ACQUIRE, __HIP_MEMORY_SCOPE_AGENT) == g) {
        __builtin_amdgcn_s_sleep(1);
      }
    }
  }
  __syncthreads();
}

// ---------------------------------------------------------------------------
// Phase 1: Xp = X @ W_xh + b    (M=32768, N=1024, K=1024), result -> Y (d_out)
// ---------------------------------------------------------------------------
__global__ void rnn_input_proj(const float* __restrict__ X,
                               const float* __restrict__ Wxh,
                               const float* __restrict__ bh,
                               float* __restrict__ Y) {
  __shared__ _Float16 ldsW[64 * KP];

  const int nblk = blockIdx.x & (HID / 64 - 1);
  const int mblk = blockIdx.x / (HID / 64);
  const int col0 = nblk * 64;
  const int row0 = mblk * 64;

  for (int i = threadIdx.x; i < 64 * IN_DIM; i += 256) {
    int n = i & 63;
    int k = i >> 6;
    ldsW[n * KP + k] = (_Float16)Wxh[(size_t)k * HID + col0 + n];
  }
  __syncthreads();

  const int lane = threadIdx.x & 31;
  const int wave = threadIdx.x >> 5;
  const int mt  = wave & 3;
  const int ntb = (wave >> 2) * 2;
  const int arow = row0 + mt * 16 + (lane & 15);
  const int kA = (lane >> 4) * 16;
  const int kB = (lane >> 4) * 16;

  const _Float16* bp0 = ldsW + ((ntb + 0) * 16 + (lane & 15)) * KP + kB;
  const _Float16* bp1 = ldsW + ((ntb + 1) * 16 + (lane & 15)) * KP + kB;
  const float* ap = X + (size_t)arow * IN_DIM + kA;

  v8f acc0 = {}, acc1 = {};
  for (int kk = 0; kk < IN_DIM; kk += 32) {
    v4f a0 = *(const v4f*)(ap + kk);
    v4f a1 = *(const v4f*)(ap + kk + 4);
    v4f a2 = *(const v4f*)(ap + kk + 8);
    v4f a3 = *(const v4f*)(ap + kk + 12);
    v16h a;
#pragma unroll
    for (int i = 0; i < 4; ++i) {
      a[i]      = (_Float16)a0[i];
      a[4 + i]  = (_Float16)a1[i];
      a[8 + i]  = (_Float16)a2[i];
      a[12 + i] = (_Float16)a3[i];
    }
    v16h b0 = *(const v16h*)(bp0 + kk);
    v16h b1 = *(const v16h*)(bp1 + kk);
    acc0 = __builtin_amdgcn_wmma_f32_16x16x32_f16(false, a, false, b0, (short)0, acc0, false, false);
    acc1 = __builtin_amdgcn_wmma_f32_16x16x32_f16(false, a, false, b1, (short)0, acc1, false, false);
  }

  const int rbase = row0 + mt * 16 + 8 * (lane >> 4);
  const int c0 = col0 + (ntb + 0) * 16 + (lane & 15);
  const int c1 = col0 + (ntb + 1) * 16 + (lane & 15);
  const float bh0 = bh[c0];
  const float bh1 = bh[c1];
#pragma unroll
  for (int r = 0; r < 8; ++r) {
    size_t row = (size_t)(rbase + r);
    Y[row * HID + c0] = acc0[r] + bh0;
    Y[row * HID + c1] = acc1[r] + bh1;
  }
}

// ---------------------------------------------------------------------------
// Phase 2: persistent recurrent scan (16 blocks x 256 threads).
// W_hh slab in LDS for all 512 steps; state ping-pongs via global f16 buffers;
// f16 state write-back staged in LDS and drained with async LDS->global stores.
// ---------------------------------------------------------------------------
__global__ void rnn_recurrent(float* __restrict__ Y,
                              const float* __restrict__ Whh,
                              _Float16* __restrict__ s0,
                              _Float16* __restrict__ s1,
                              unsigned* __restrict__ bar) {
  __shared__ _Float16 ldsW[64 * KP];
  __shared__ _Float16 ldsS[64 * 64];   // h_t slab staging (rows x local cols)

  const int col0 = blockIdx.x * 64;

  for (int i = threadIdx.x; i < 64 * HID; i += 256) {
    int n = i & 63;
    int k = i >> 6;
    ldsW[n * KP + k] = (_Float16)Whh[(size_t)k * HID + col0 + n];
  }
  __syncthreads();

  const int lane = threadIdx.x & 31;
  const int wave = threadIdx.x >> 5;
  const int mt  = wave & 3;
  const int ntb = (wave >> 2) * 2;
  const int arow = mt * 16 + (lane & 15);
  const int kA = (lane >> 4) * 16;
  const int kB = (lane >> 4) * 16;

  const _Float16* bp0 = ldsW + ((ntb + 0) * 16 + (lane & 15)) * KP + kB;
  const _Float16* bp1 = ldsW + ((ntb + 1) * 16 + (lane & 15)) * KP + kB;

  const int rbase = mt * 16 + 8 * (lane >> 4);
  const int lc0 = (ntb + 0) * 16 + (lane & 15);   // local col in [0,64)
  const int lc1 = (ntb + 1) * 16 + (lane & 15);
  const int c0 = col0 + lc0;
  const int c1 = col0 + lc1;

  for (int t = 0; t < T_STEPS; ++t) {
    const _Float16* sRead = (t & 1) ? s1 : s0;   // holds h_{t-1}
    _Float16*       sWrite = (t & 1) ? s0 : s1;  // receives h_t

    const _Float16* ap = sRead + (size_t)arow * HID + kA;

    v8f acc0 = {}, acc1 = {};
#pragma unroll 4
    for (int kk = 0; kk < HID; kk += 32) {
      v8h alo = *(const v8h*)(ap + kk);
      v8h ahi = *(const v8h*)(ap + kk + 8);
      v16h a = __builtin_shufflevector(alo, ahi, 0, 1, 2, 3, 4, 5, 6, 7,
                                       8, 9, 10, 11, 12, 13, 14, 15);
      v16h b0 = *(const v16h*)(bp0 + kk);
      v16h b1 = *(const v16h*)(bp1 + kk);
      acc0 = __builtin_amdgcn_wmma_f32_16x16x32_f16(false, a, false, b0, (short)0, acc0, false, false);
      acc1 = __builtin_amdgcn_wmma_f32_16x16x32_f16(false, a, false, b1, (short)0, acc1, false, false);
    }

    float* yrow = Y + (size_t)t * BATCH * HID;
#pragma unroll
    for (int r = 0; r < 8; ++r) {
      size_t row = (size_t)(rbase + r);
      float h0 = FAST_TANH(yrow[row * HID + c0] + acc0[r]);
      float h1 = FAST_TANH(yrow[row * HID + c1] + acc1[r]);
      yrow[row * HID + c0] = h0;
      yrow[row * HID + c1] = h1;
#if defined(HAVE_ASYNC_ST)
      ldsS[row * 64 + lc0] = (_Float16)h0;
      ldsS[row * 64 + lc1] = (_Float16)h1;
#else
      sWrite[row * HID + c0] = (_Float16)h0;
      sWrite[row * HID + c1] = (_Float16)h1;
#endif
      if (t == T_STEPS - 1) {  // final_state appended after outputs
        Y[(size_t)T_STEPS * BATCH * HID + row * HID + c0] = h0;
        Y[(size_t)T_STEPS * BATCH * HID + row * HID + c1] = h1;
      }
    }

#if defined(HAVE_ASYNC_ST)
    __syncthreads();  // ldsS fully written
    // Drain 8 KB staging tile: 512 chunks of 16B; every lane moves 16B per call.
    for (int i = threadIdx.x; i < 512; i += 256) {
      int row = i >> 3;
      int off = (i & 7) * 8;            // f16 elements within the 64-wide row
      __builtin_amdgcn_global_store_async_from_lds_b128(
          (__attribute__((address_space(1))) i32x4*)(sWrite + (size_t)row * HID + col0 + off),
          (__attribute__((address_space(3))) i32x4*)(ldsS + row * 64 + off),
          0, 0);
    }
    wait_asynccnt0();
#endif

    grid_barrier(bar, NBLK2);
  }
}

// ---------------------------------------------------------------------------
extern "C" void kernel_launch(void* const* d_in, const int* in_sizes, int n_in,
                              void* d_out, int out_size, void* d_ws, size_t ws_size,
                              hipStream_t stream) {
  const float* X   = (const float*)d_in[0];   // [T,B,I]
  const float* Wxh = (const float*)d_in[1];   // [I,H]
  const float* Whh = (const float*)d_in[2];   // [H,H]
  const float* bh  = (const float*)d_in[3];   // [H]
  float* Y = (float*)d_out;                   // [T,B,H] outputs ++ [B,H] final state

  _Float16* s0 = (_Float16*)d_ws;
  _Float16* s1 = s0 + (size_t)BATCH * HID;
  unsigned* bar = (unsigned*)((char*)d_ws + 2u * BATCH * HID * sizeof(_Float16));

  rnn_init<<<64, 256, 0, stream>>>(s0, s1, bar);

  const int grid1 = (T_STEPS * BATCH / 64) * (HID / 64);  // 8192
  rnn_input_proj<<<grid1, 256, 0, stream>>>(X, Wxh, bh, Y);

  rnn_recurrent<<<NBLK2, 256, 0, stream>>>(Y, Whh, s0, s1, bar);
}
